// GraphBlock_63917703299231
// MI455X (gfx1250) — compile-verified, matrix operand order checked by reading, not run on hardware
//
#include <hip/hip_runtime.h>
#include <math.h>

#define N_ 16
#define P_ 24
#define S_ 160
#define ENT_ 263
#define E_ 448
#define D_ 1024
#define KH 8
#define T_ 3
#define O_ 128
#define HDIM 2048
#define NEGV -1e30f
#define TKSTR 57344L  /* KH*N_*E_ : t-stride in left/right */

// ---- d_out layout (float offsets, reference return order) ----
#define O_GS     0L
#define O_NPARA  7340032L
#define O_NSENT  7733248L
#define O_NENT   14663680L
#define O_NM     18972672L
#define O_QOUT   18979840L
#define O_PLOG   18996224L
#define O_PPRED  18996608L
#define O_SLOG   18997376L
#define O_SPRED  18999936L
#define O_ELOG   19005056L

// ---- workspace layout (float offsets; bf16 regions cast) ----
#define W_GS0   0L         /* fp32 graph_state 16*448*1024            */
#define W_WL    7340032L
#define W_WR    7733248L
#define W_LEFT  8126464L
#define W_RIGHT 8298496L
#define W_MX    8470528L
#define W_RDEN  8527872L
#define W_HM    8585216L   /* bf16 packed-B hm: 8*16*448*128 bf16     */
#define W_HS    12255232L  /* fp32 2960*2048                          */
#define W_HE    18317312L  /* fp32 4208*2048                          */
#define W_GS0H  26935296L  /* bf16 graph_state shadow                 */
#define W_GSH   30605312L  /* bf16 gat-output shadow                  */
#define W_PGLW  34275328L  /* packed bf16 gat_lin_W  1024x1024        */
#define W_PWT2  34799616L  /* packed bf16 W_type[2]  8 x 1024x128     */
#define W_PSW1  35323904L  /* packed bf16 sent_W1    1024x2048        */
#define W_PEW1  36372480L  /* packed bf16 ent_W1     1024x2048        */

typedef __attribute__((ext_vector_type(16))) __bf16 v16bf;
typedef __attribute__((ext_vector_type(8)))  __bf16 v8bf;
typedef __attribute__((ext_vector_type(8)))  float  v8f;
typedef __attribute__((ext_vector_type(4)))  float  v4f;

static __device__ __forceinline__ v8f wmma_bf16(v16bf a, v16bf b, v8f c) {
  return __builtin_amdgcn_wmma_f32_16x16x32_bf16(false, a, false, b, (short)0, c, false, false);
}

static __device__ __forceinline__ v8f zero8() {
  v8f z;
#pragma unroll
  for (int i = 0; i < 8; ++i) z[i] = 0.f;
  return z;
}

// A fragment (16x32, lane(l,hi): row m=l, K elems {0..7,16..23}+8*hi) from fp32 row
static __device__ __forceinline__ v16bf frag_a_f32(const float* __restrict__ row, int k0, int hi) {
  const float* r = row + k0 + 8 * hi;
  v4f x0 = *(const v4f*)(r);
  v4f x1 = *(const v4f*)(r + 4);
  v4f x2 = *(const v4f*)(r + 16);
  v4f x3 = *(const v4f*)(r + 20);
  v16bf a;
#pragma unroll
  for (int i = 0; i < 4; ++i) {
    a[i]      = (__bf16)x0[i];
    a[4 + i]  = (__bf16)x1[i];
    a[8 + i]  = (__bf16)x2[i];
    a[12 + i] = (__bf16)x3[i];
  }
  return a;
}

// A fragment from bf16 row-major (two b128 loads)
static __device__ __forceinline__ v16bf frag_a_bf16(const __bf16* __restrict__ row, int k0, int hi) {
  const __bf16* r = row + k0 + 8 * hi;
  v8bf lo = *(const v8bf*)(r);
  v8bf hh = *(const v8bf*)(r + 16);
  return __builtin_shufflevector(lo, hh, 0, 1, 2, 3, 4, 5, 6, 7, 8, 9, 10, 11, 12, 13, 14, 15);
}

// B fragment from packed tiles: tile*512 + lane*16, contiguous (two b128 loads)
static __device__ __forceinline__ v16bf frag_b_packed(const __bf16* __restrict__ packed,
                                                      long tile, int lane) {
  const __bf16* p = packed + (tile * 32 + lane) * 16;
  v8bf lo = *(const v8bf*)(p);
  v8bf hh = *(const v8bf*)(p + 8);
  return __builtin_shufflevector(lo, hh, 0, 1, 2, 3, 4, 5, 6, 7, 8, 9, 10, 11, 12, 13, 14, 15);
}

static __device__ __forceinline__ float block_sum(float v, float* buf) {
  int tid = threadIdx.x;
  buf[tid] = v; __syncthreads();
  for (int s = 128; s > 0; s >>= 1) {
    if (tid < s) buf[tid] += buf[tid + s];
    __syncthreads();
  }
  float r = buf[0]; __syncthreads();
  return r;
}

// pack fp32 KxN weight -> bf16 fragment-native tiles
__global__ void k_pack_b(const float* __restrict__ B, __bf16* __restrict__ out,
                         int Kdim, int Nn) {
  long idx = (long)blockIdx.x * blockDim.x + threadIdx.x;
  if (idx >= (long)Kdim * Nn) return;
  int e = (int)(idx & 15);
  int lane = (int)((idx >> 4) & 31);
  long tile = idx >> 9;
  int NT = Nn >> 4;
  int nt = (int)(tile % NT);
  int kt = (int)(tile / NT);
  int l = lane & 15, hi = lane >> 4;
  int p = e >> 1, q = e & 1;
  int kb = ((p < 4) ? 2 * p : 16 + 2 * (p - 4)) + 8 * hi;
  int k = kt * 32 + kb + q;
  int n = nt * 16 + l;
  out[idx] = (__bf16)B[(long)k * Nn + n];
}

// q = query @ gat_lin_W + b -> graph_state row 0 (one block: 8 waves x 8 n-tiles)
__global__ void k_q(const float* __restrict__ query, const __bf16* __restrict__ Wp,
                    const float* __restrict__ bias, float* __restrict__ gs0,
                    __bf16* __restrict__ gs0h) {
  int wave = threadIdx.x >> 5;
  int lane = threadIdx.x & 31, l = lane & 15, hi = lane >> 4;
  const float* arow = query + (long)l * D_;
  v8f acc[8];
#pragma unroll
  for (int t = 0; t < 8; ++t) acc[t] = zero8();
  for (int k0 = 0; k0 < D_; k0 += 32) {
    v16bf a = frag_a_f32(arow, k0, hi);
    int kt = k0 >> 5;
#pragma unroll
    for (int t = 0; t < 8; ++t) {
      v16bf b = frag_b_packed(Wp, (long)kt * 64 + wave * 8 + t, lane);
      acc[t] = wmma_bf16(a, b, acc[t]);
    }
  }
#pragma unroll
  for (int t = 0; t < 8; ++t) {
    int n0 = (wave * 8 + t) * 16;
#pragma unroll
    for (int r = 0; r < 8; ++r) {
      int m = r + 8 * hi;  // batch index
      float v = acc[t][r] + bias[n0 + l];
      gs0[((long)m * E_) * D_ + n0 + l] = v;
      gs0h[((long)m * E_) * D_ + n0 + l] = (__bf16)v;
    }
  }
}

// concat(para,sent,ent) into graph_state rows 1..447 (fp32 + bf16) + node_mask out
__global__ void k_assemble(const float* __restrict__ para, const float* __restrict__ sent,
                           const float* __restrict__ ent, const float* __restrict__ pm,
                           const float* __restrict__ sm, const float* __restrict__ em,
                           float* __restrict__ gs0, __bf16* __restrict__ gs0h,
                           float* __restrict__ dout) {
  long idx = (long)blockIdx.x * blockDim.x + threadIdx.x;
  if (idx < (long)N_ * E_) {
    int e = (int)(idx % E_), n = (int)(idx / E_);
    float m;
    if (e == 0) m = 1.f;
    else if (e < 1 + P_) m = pm[n * P_ + e - 1];
    else if (e < 1 + P_ + S_) m = sm[n * S_ + e - 1 - P_];
    else m = em[n * ENT_ + e - 1 - P_ - S_];
    dout[O_NM + idx] = m;
  }
  long tot = (long)N_ * (E_ - 1) * D_;
  if (idx >= tot) return;
  int d = (int)(idx % D_);
  long r = idx / D_;
  int e = (int)(r % (E_ - 1)) + 1;
  int n = (int)(r / (E_ - 1));
  float v;
  if (e < 1 + P_) v = para[((long)n * P_ + e - 1) * D_ + d];
  else if (e < 1 + P_ + S_) v = sent[((long)n * S_ + e - 1 - P_) * D_ + d];
  else v = ent[((long)n * ENT_ + e - 1 - P_ - S_) * D_ + d];
  long o = ((long)n * E_ + e) * D_ + d;
  gs0[o] = v;
  gs0h[o] = (__bf16)v;
}

// gates: q1=relu(q@qW1), g=sigmoid(q1@qW2); wl/wr = W_type @ (g*a) halves
__global__ void k_gate(const float* __restrict__ query, const float* __restrict__ qW1,
                       const float* __restrict__ qW2, const float* __restrict__ at,
                       const float* __restrict__ Wt, float* __restrict__ wl,
                       float* __restrict__ wr) {
  __shared__ float q1[256];
  __shared__ float gl[O_], gr[O_];
  int b = blockIdx.x;
  int n = b % N_, tk = b / N_;
  int tid = threadIdx.x;
  const float* qn = query + (long)n * D_;
  const float* w1 = qW1 + (long)tk * D_ * 256;
  float s = 0.f;
  for (int d = 0; d < D_; ++d) s += qn[d] * w1[(long)d * 256 + tid];
  q1[tid] = fmaxf(s, 0.f);
  __syncthreads();
  const float* w2 = qW2 + (long)tk * 256 * 256;
  float s2 = 0.f;
  for (int m = 0; m < 256; ++m) s2 += q1[m] * w2[m * 256 + tid];
  float g = 1.f / (1.f + __expf(-s2));
  float av = at[tk * 256 + tid];
  if (tid < O_) gl[tid] = g * av; else gr[tid - O_] = g * av;
  __syncthreads();
  const float* W = Wt + (long)tk * D_ * O_;
  for (int d = tid; d < D_; d += 256) {
    const float* row = W + (long)d * O_;
    float sl = 0.f, sr = 0.f;
    for (int o = 0; o < O_; ++o) { sl += row[o] * gl[o]; sr += row[o] * gr[o]; }
    long base = ((long)tk * N_ + n) * D_ + d;
    wl[base] = sl; wr[base] = sr;
  }
}

// left/right[t,k,n,e] = gs0[n,e,:] . wl/wr[t,k,n,:]  (one wave per (tk,n,e))
__global__ void k_lr(const float* __restrict__ gs0, const float* __restrict__ wl,
                     const float* __restrict__ wr, float* __restrict__ left,
                     float* __restrict__ right) {
  long gw = ((long)blockIdx.x * blockDim.x + threadIdx.x) >> 5;
  int lane = threadIdx.x & 31;
  int e = (int)(gw % E_); long r = gw / E_;
  int n = (int)(r % N_); int tk = (int)(r / N_);
  const float* x = gs0 + ((long)n * E_ + e) * D_;
  const float* a = wl + ((long)tk * N_ + n) * D_;
  const float* b = wr + ((long)tk * N_ + n) * D_;
  float sl = 0.f, sr = 0.f;
  for (int d = lane; d < D_; d += 32) { float xv = x[d]; sl += xv * a[d]; sr += xv * b[d]; }
#pragma unroll
  for (int off = 16; off > 0; off >>= 1) {
    sl += __shfl_down(sl, off, 32);
    sr += __shfl_down(sr, off, 32);
  }
  if (lane == 0) {
    long o = ((long)tk * N_ + n) * E_ + e;
    left[o] = sl; right[o] = sr;
  }
}

// hm[k,n] = (gs[n] @ W_type[2,k]) * mask, written in packed-B fragment layout (bf16)
__global__ void k_h2(const __bf16* __restrict__ gs0h, const __bf16* __restrict__ Wp2,
                     const float* __restrict__ nm, __bf16* __restrict__ hmp) {
  long gw = ((long)blockIdx.x * blockDim.x + threadIdx.x) >> 5;
  int lane = threadIdx.x & 31, l = lane & 15, hi = lane >> 4;
  int mt = (int)(gw % 28); long r = gw / 28;
  int n = (int)(r % N_); int k = (int)(r / N_);
  const __bf16* arow = gs0h + ((long)n * E_ + mt * 16 + l) * D_;
  const __bf16* Bp = Wp2 + (long)k * D_ * O_;
  v8f acc[8];
#pragma unroll
  for (int t = 0; t < 8; ++t) acc[t] = zero8();
  for (int k0 = 0; k0 < D_; k0 += 32) {
    v16bf a = frag_a_bf16(arow, k0, hi);
    int kt = k0 >> 5;
#pragma unroll
    for (int ot = 0; ot < 8; ++ot) {
      v16bf b = frag_b_packed(Bp, (long)kt * 8 + ot, lane);
      acc[ot] = wmma_bf16(a, b, acc[ot]);
    }
  }
  long base = ((long)k * N_ + n) * (14L * 8 * 512);
#pragma unroll
  for (int ot = 0; ot < 8; ++ot) {
#pragma unroll
    for (int rr = 0; rr < 8; ++rr) {
      int i = mt * 16 + rr + 8 * hi;              // K index of the aggr GEMM
      int it = i >> 5, kb = i & 31;
      int b2 = kb >> 1, q = kb & 1;
      int hi_r = (b2 >> 2) & 1;
      int p = (b2 & 3) + ((b2 >> 3) << 2);
      int e_idx = 2 * p + q;
      int lane_r = hi_r * 16 + l;                 // l is the o-column low bits
      float v = acc[ot][rr] * nm[n * E_ + i];
      hmp[base + ((long)it * 8 + ot) * 512 + lane_r * 16 + e_idx] = (__bf16)v;
    }
  }
}

// softmax pass 1: row max + 1/sum over j (scores recomputed from left/right/adj)
__global__ void k_stats(const int* __restrict__ adj, const float* __restrict__ left,
                        const float* __restrict__ right, float* __restrict__ mx,
                        float* __restrict__ rden) {
  long gw = ((long)blockIdx.x * blockDim.x + threadIdx.x) >> 5;
  int lane = threadIdx.x & 31;
  int i = (int)(gw % E_); long r = gw / E_;
  int n = (int)(r % N_); int k = (int)(r / N_);
  long knE = ((long)k * N_ + n) * E_;
  float l0 = left[knE + i], l1 = left[knE + TKSTR + i], l2 = left[knE + 2 * TKSTR + i];
  const int* arow = adj + ((long)n * E_ + i) * E_;
  float s[14];
  float m = NEGV;
#pragma unroll
  for (int jj = 0; jj < 14; ++jj) {
    int j = lane + 32 * jj;
    int av = arow[j];
    float v = NEGV;
    if (av > 0) {
      float lf = (av == 1) ? l0 : ((av == 2) ? l1 : l2);
      float x = lf + right[knE + (long)(av - 1) * TKSTR + j];
      v = (x > 0.f) ? x : 0.2f * x;
    }
    s[jj] = v;
    m = fmaxf(m, v);
  }
#pragma unroll
  for (int d = 16; d > 0; d >>= 1) m = fmaxf(m, __shfl_xor(m, d, 32));
  float den = 0.f;
#pragma unroll
  for (int jj = 0; jj < 14; ++jj) den += __expf(s[jj] - m);
#pragma unroll
  for (int d = 16; d > 0; d >>= 1) den += __shfl_xor(den, d, 32);
  if (lane == 0) { mx[knE + i] = m; rden[knE + i] = 1.f / den; }
}

// pass 2: coef A-fragments built once per wave, used for all 8 o-tiles (8 WMMAs/step)
__global__ void k_aggr(const int* __restrict__ adj, const float* __restrict__ left,
                       const float* __restrict__ right, const float* __restrict__ mx,
                       const float* __restrict__ rden, const __bf16* __restrict__ hmp,
                       float* __restrict__ gs, __bf16* __restrict__ gsh) {
  long gw = ((long)blockIdx.x * blockDim.x + threadIdx.x) >> 5;
  int lane = threadIdx.x & 31, l = lane & 15, hi = lane >> 4;
  int jt = (int)(gw % 28); long r = gw / 28;
  int n = (int)(r % N_); int k = (int)(r / N_);
  int j = jt * 16 + l;
  long knE = ((long)k * N_ + n) * E_;
  float rj0 = right[knE + j];
  float rj1 = right[knE + TKSTR + j];
  float rj2 = right[knE + 2 * TKSTR + j];
  const int* adjn = adj + (long)n * E_ * E_;
  const float* mxp = mx + knE;
  const float* rdp = rden + knE;
  const __bf16* Bp = hmp + ((long)k * N_ + n) * (14L * 8 * 512);
  v8f acc[8];
#pragma unroll
  for (int t = 0; t < 8; ++t) acc[t] = zero8();
  for (int i0 = 0; i0 < E_; i0 += 32) {
    v16bf a;
#pragma unroll
    for (int p = 0; p < 8; ++p) {
      int kb = ((p < 4) ? 2 * p : 16 + 2 * (p - 4)) + 8 * hi;
#pragma unroll
      for (int q = 0; q < 2; ++q) {
        int i = i0 + kb + q;
        int av = adjn[(long)i * E_ + j];
        float sv = NEGV;
        if (av > 0) {
          float lf = left[knE + (long)(av - 1) * TKSTR + i];
          float rv = (av == 1) ? rj0 : ((av == 2) ? rj1 : rj2);
          float x = lf + rv;
          sv = (x > 0.f) ? x : 0.2f * x;
        }
        float coef = __expf(sv - mxp[i]) * rdp[i];
        a[2 * p + q] = (__bf16)coef;
      }
    }
    int it = i0 >> 5;
#pragma unroll
    for (int ot = 0; ot < 8; ++ot) {
      v16bf b = frag_b_packed(Bp, (long)it * 8 + ot, lane);
      acc[ot] = wmma_bf16(a, b, acc[ot]);
    }
  }
#pragma unroll
  for (int ot = 0; ot < 8; ++ot) {
#pragma unroll
    for (int rr = 0; rr < 8; ++rr) {
      int jj = jt * 16 + rr + 8 * hi;
      float v = fmaxf(acc[ot][rr], 0.f);
      long o = ((long)n * E_ + jj) * D_ + k * O_ + ot * 16 + l;
      gs[o] = v;
      gsh[o] = (__bf16)v;
    }
  }
}

// H = relu(X @ W1 + b1); one wave handles 8 N-tiles (A fragment amortized)
__global__ void k_gemm_out(const __bf16* __restrict__ gsh, const __bf16* __restrict__ W1p,
                           const float* __restrict__ b1, float* __restrict__ H,
                           int rows, int ebase) {
  long gw = ((long)blockIdx.x * blockDim.x + threadIdx.x) >> 5;
  int lane = threadIdx.x & 31, l = lane & 15, hi = lane >> 4;
  int ng = (int)(gw % 16);            // group of 8 n-tiles (HDIM/16/8 = 16 groups)
  int mt = (int)(gw / 16);
  int m = mt * 16 + l;
  int n = m / rows, e = ebase + m % rows;
  const __bf16* arow = gsh + ((long)n * E_ + e) * D_;
  v8f acc[8];
#pragma unroll
  for (int t = 0; t < 8; ++t) acc[t] = zero8();
  for (int k0 = 0; k0 < D_; k0 += 32) {
    v16bf a = frag_a_bf16(arow, k0, hi);
    int kt = k0 >> 5;
#pragma unroll
    for (int t = 0; t < 8; ++t) {
      v16bf b = frag_b_packed(W1p, (long)kt * 128 + ng * 8 + t, lane);
      acc[t] = wmma_bf16(a, b, acc[t]);
    }
  }
#pragma unroll
  for (int t = 0; t < 8; ++t) {
    int n0 = (ng * 8 + t) * 16;
#pragma unroll
    for (int rr = 0; rr < 8; ++rr) {
      int mm = mt * 16 + rr + 8 * hi;
      H[(long)mm * HDIM + n0 + l] = fmaxf(acc[t][rr] + b1[n0 + l], 0.f);
    }
  }
}

// LayerNorm + dot(W2) + b2, route logits into d_out slots
__global__ void k_lndot(const float* __restrict__ H, const float* __restrict__ g,
                        const float* __restrict__ beta, const float* __restrict__ W2,
                        const float* __restrict__ b2, const float* __restrict__ acm,
                        float* __restrict__ dout, int rows, int is_ent) {
  __shared__ float buf[256];
  int row = blockIdx.x, tid = threadIdx.x;
  const float* h = H + (long)row * HDIM;
  float hv[8];
  float s = 0.f, sq = 0.f;
#pragma unroll
  for (int i = 0; i < 8; ++i) {
    float v = h[tid + i * 256];
    hv[i] = v; s += v; sq += v * v;
  }
  s = block_sum(s, buf);
  sq = block_sum(sq, buf);
  float mean = s * (1.f / HDIM);
  float var = sq * (1.f / HDIM) - mean * mean;
  float rstd = rsqrtf(var + 1e-12f);
  float dot = 0.f;
#pragma unroll
  for (int i = 0; i < 8; ++i) {
    int cidx = tid + i * 256;
    float hn = (hv[i] - mean) * rstd * g[cidx] + beta[cidx];
    dot += hn * W2[cidx];
  }
  dot = block_sum(dot, buf);
  if (tid == 0) {
    float v = dot + b2[0];
    int n = row / rows, e = row % rows;
    if (is_ent) {
      dout[O_ELOG + n * ENT_ + e] = v - 1e30f * (1.f - acm[n * ENT_ + e]);
    } else if (e >= 1 && e <= P_) {
      int pi = n * P_ + (e - 1);
      dout[O_PLOG + pi] = v;
      dout[O_PPRED + 2 * pi] = 0.f;
      dout[O_PPRED + 2 * pi + 1] = v;
    } else if (e > P_) {
      int si = n * S_ + (e - 1 - P_);
      dout[O_SLOG + si] = v;
      dout[O_SPRED + 2 * si] = 0.f;
      dout[O_SPRED + 2 * si + 1] = v;
    }
  }
}

// copy gs[:, e0:e0+cnt, :] -> dst
__global__ void k_slice(const float* __restrict__ gs, float* __restrict__ dst,
                        int e0, int cnt) {
  long idx = (long)blockIdx.x * blockDim.x + threadIdx.x;
  long tot = (long)N_ * cnt * D_;
  if (idx >= tot) return;
  int d = (int)(idx % D_);
  long r = idx / D_;
  int e = (int)(r % cnt);
  int n = (int)(r / cnt);
  dst[idx] = gs[((long)n * E_ + e0 + e) * D_ + d];
}

extern "C" void kernel_launch(void* const* d_in, const int* in_sizes, int n_in,
                              void* d_out, int out_size, void* d_ws, size_t ws_size,
                              hipStream_t stream) {
  const float* para  = (const float*)d_in[0];
  const float* sentp = (const float*)d_in[1];
  const float* entp  = (const float*)d_in[2];
  const float* query = (const float*)d_in[3];
  const float* pm    = (const float*)d_in[4];
  const float* sm    = (const float*)d_in[5];
  const float* em    = (const float*)d_in[6];
  const float* acm   = (const float*)d_in[7];
  const int*   adj   = (const int*)d_in[8];
  const float* glW   = (const float*)d_in[9];
  const float* glb   = (const float*)d_in[10];
  const float* Wt    = (const float*)d_in[11];
  const float* at    = (const float*)d_in[12];
  const float* qW1   = (const float*)d_in[13];
  const float* qW2   = (const float*)d_in[14];
  const float* sW1   = (const float*)d_in[15];
  const float* sb1   = (const float*)d_in[16];
  const float* sg    = (const float*)d_in[17];
  const float* sbeta = (const float*)d_in[18];
  const float* sW2   = (const float*)d_in[19];
  const float* sb2   = (const float*)d_in[20];
  const float* eW1   = (const float*)d_in[21];
  const float* eb1   = (const float*)d_in[22];
  const float* eg    = (const float*)d_in[23];
  const float* ebeta = (const float*)d_in[24];
  const float* eW2   = (const float*)d_in[25];
  const float* eb2   = (const float*)d_in[26];
  float* out = (float*)d_out;
  float* ws  = (float*)d_ws;

  float*  gs0   = ws + W_GS0;
  float*  wl    = ws + W_WL;
  float*  wr    = ws + W_WR;
  float*  left  = ws + W_LEFT;
  float*  right = ws + W_RIGHT;
  float*  mx    = ws + W_MX;
  float*  rden  = ws + W_RDEN;
  __bf16* hmp   = (__bf16*)(ws + W_HM);
  float*  Hs    = ws + W_HS;
  float*  He    = ws + W_HE;
  __bf16* gs0h  = (__bf16*)(ws + W_GS0H);
  __bf16* gsh   = (__bf16*)(ws + W_GSH);
  __bf16* pGlW  = (__bf16*)(ws + W_PGLW);
  __bf16* pWt2  = (__bf16*)(ws + W_PWT2);
  __bf16* pSW1  = (__bf16*)(ws + W_PSW1);
  __bf16* pEW1  = (__bf16*)(ws + W_PEW1);

  float* gs = out + O_GS;
  float* nm = out + O_NM;

  // pack reused B operands to bf16 fragment tiles
  k_pack_b<<<4096, 256, 0, stream>>>(glW, pGlW, D_, D_);
  for (int k = 0; k < KH; ++k)
    k_pack_b<<<512, 256, 0, stream>>>(Wt + (long)(2 * KH + k) * D_ * O_,
                                      pWt2 + (long)k * D_ * O_, D_, O_);
  k_pack_b<<<8192, 256, 0, stream>>>(sW1, pSW1, D_, HDIM);
  k_pack_b<<<8192, 256, 0, stream>>>(eW1, pEW1, D_, HDIM);

  k_q<<<1, 256, 0, stream>>>(query, pGlW, glb, gs0, gs0h);
  k_assemble<<<28608, 256, 0, stream>>>(para, sentp, entp, pm, sm, em, gs0, gs0h, out);
  k_gate<<<T_ * KH * N_, 256, 0, stream>>>(query, qW1, qW2, at, Wt, wl, wr);
  k_lr<<<21504, 256, 0, stream>>>(gs0, wl, wr, left, right);
  k_h2<<<448, 256, 0, stream>>>(gs0h, pWt2, nm, hmp);
  k_stats<<<7168, 256, 0, stream>>>(adj, left, right, mx, rden);
  k_aggr<<<448, 256, 0, stream>>>(adj, left, right, mx, rden, hmp, gs, gsh);

  k_slice<<<1536, 256, 0, stream>>>(gs, out + O_NPARA, 1, P_);
  k_slice<<<27072, 256, 0, stream>>>(gs, out + O_NSENT, 1 + P_, E_ - 1 - P_);
  k_slice<<<16832, 256, 0, stream>>>(gs, out + O_NENT, 1 + P_ + S_, ENT_);
  k_slice<<<64, 256, 0, stream>>>(gs, out + O_QOUT, 0, 1);

  k_gemm_out<<<370, 256, 0, stream>>>(gsh, pSW1, sb1, Hs, 1 + P_ + S_, 0);
  k_gemm_out<<<526, 256, 0, stream>>>(gsh, pEW1, eb1, He, ENT_, 1 + P_ + S_);
  k_lndot<<<2960, 256, 0, stream>>>(Hs, sg, sbeta, sW2, sb2, acm, out, 1 + P_ + S_, 0);
  k_lndot<<<4208, 256, 0, stream>>>(He, eg, ebeta, eW2, eb2, acm, out, ENT_, 1);

  (void)in_sizes; (void)n_in; (void)out_size; (void)ws_size;
}
